// DGCNLayer_cross_50560355009139
// MI455X (gfx1250) — compile-verified, matrix operand order checked by reading, not run on hardware
//
#include <hip/hip_runtime.h>
#include <hip/hip_bf16.h>

#define F_DIM      256
#define N_USERS_C  100000
#define N_ITEMS_C  50000
#define NNZ_C      800000
#define NTW        4            // N-tiles (of 16) per wave -> 16x64 strip

typedef __attribute__((ext_vector_type(16))) __bf16 v16bf;
typedef __attribute__((ext_vector_type(8)))  float  v8f;

// ---------------------------------------------------------------------------
// Pack weight B[K x N] (row-major, ldb=N) into per-lane WMMA-B layout, bf16
// hi/lo split.  One wave per (n-tile, k-step).  Block layout per (tn, ks):
//   1024 bf16 = [ hi: 32 lanes x 16 elems | lo: 32 lanes x 16 elems ]
// ---------------------------------------------------------------------------
__global__ __launch_bounds__(32)
void pack_b_kernel(const float* __restrict__ B, int ldb, int Ksteps,
                   __bf16* __restrict__ Bp) {
    const int g    = blockIdx.x;
    const int tn   = g / Ksteps;
    const int ks   = g % Ksteps;
    const int lane = threadIdx.x;
    const int half = lane >> 4;
    const int l15  = lane & 15;
    const int k0   = ks << 5;
    __bf16* dst = Bp + ((long long)tn * Ksteps + ks) * 1024 + lane * 16;
#pragma unroll
    for (int j = 0; j < 16; ++j) {
        const int k = k0 + ((j >> 3) << 4) + (half << 3) + (j & 7);
        const float x = B[(long long)k * ldb + tn * 16 + l15];
        const __bf16 h = (__bf16)x;
        dst[j]       = h;
        dst[512 + j] = (__bf16)(x - (float)h);
    }
}

// ---------------------------------------------------------------------------
// GEMM: C[M,N] = concat_K(A1[M,K1], A2[M,K2]) @ B[(K1+K2), N]
// One wave computes a 16x64 strip (4 accumulators).  A loaded as 4x b128 and
// hi/lo-split once per K-step; B read pre-packed (no conversion, 2x 32B/tile).
// 12 v_wmma_f32_16x16x32_bf16 per K-step per wave (hi*hi + hi*lo + lo*hi).
// ---------------------------------------------------------------------------
__global__ __launch_bounds__(32)
void gemm_bf16x3_kernel(const float* __restrict__ A1, int lda1, int K1,
                        const float* __restrict__ A2, int lda2, int K2,
                        const __bf16* __restrict__ Bp,
                        float* __restrict__ C, int ldc, int NtGroups) {
    const int g    = blockIdx.x;
    const int tm   = g / NtGroups;
    const int tn0  = (g % NtGroups) * NTW;
    const int lane = threadIdx.x;
    const int half = lane >> 4;
    const int l15  = lane & 15;
    const int m0   = tm * 16;
    const int K    = K1 + K2;
    const int Ksteps = K >> 5;

    v8f acc[NTW];
#pragma unroll
    for (int t = 0; t < NTW; ++t)
        acc[t] = (v8f){0.f, 0.f, 0.f, 0.f, 0.f, 0.f, 0.f, 0.f};

    const int arow = m0 + l15;

    for (int ks = 0; ks < Ksteps; ++ks) {
        const int k0 = ks << 5;
        // Each 32-wide K window lies entirely within A1 or A2 (K1 % 32 == 0).
        const float* Ap = (k0 < K1)
            ? (A1 + (long long)arow * lda1 + k0)
            : (A2 + (long long)arow * lda2 + (k0 - K1));
        const float4 f0 = *(const float4*)(Ap + (half << 3));
        const float4 f1 = *(const float4*)(Ap + (half << 3) + 4);
        const float4 f2 = *(const float4*)(Ap + 16 + (half << 3));
        const float4 f3 = *(const float4*)(Ap + 16 + (half << 3) + 4);
        const float af[16] = {f0.x, f0.y, f0.z, f0.w, f1.x, f1.y, f1.z, f1.w,
                              f2.x, f2.y, f2.z, f2.w, f3.x, f3.y, f3.z, f3.w};
        v16bf a_hi, a_lo;
#pragma unroll
        for (int j = 0; j < 16; ++j) {
            const __bf16 h = (__bf16)af[j];
            a_hi[j] = h;
            a_lo[j] = (__bf16)(af[j] - (float)h);
        }
#pragma unroll
        for (int t = 0; t < NTW; ++t) {
            const __bf16* bp =
                Bp + ((long long)(tn0 + t) * Ksteps + ks) * 1024 + lane * 16;
            const v16bf b_hi = *(const v16bf*)bp;
            const v16bf b_lo = *(const v16bf*)(bp + 512);
            acc[t] = __builtin_amdgcn_wmma_f32_16x16x32_bf16(
                false, a_hi, false, b_hi, (short)0, acc[t], false, false);
            acc[t] = __builtin_amdgcn_wmma_f32_16x16x32_bf16(
                false, a_hi, false, b_lo, (short)0, acc[t], false, false);
            acc[t] = __builtin_amdgcn_wmma_f32_16x16x32_bf16(
                false, a_lo, false, b_hi, (short)0, acc[t], false, false);
        }
    }

#pragma unroll
    for (int t = 0; t < NTW; ++t) {
#pragma unroll
        for (int e = 0; e < 8; ++e) {
            const int row = m0 + (half << 3) + e;
            C[(long long)row * ldc + (tn0 + t) * 16 + l15] = acc[t][e];
        }
    }
}

// ---------------------------------------------------------------------------
// SpMM scatter: Y[rows[e], :] += vals[e] * X[cols[e], :]   (Y pre-zeroed)
// 64 threads per edge, float4 per thread, atomic f32 adds (L2-resident).
// ---------------------------------------------------------------------------
__global__ void spmm_scatter_kernel(const int* __restrict__ rows,
                                    const int* __restrict__ cols,
                                    const float* __restrict__ vals,
                                    const float* __restrict__ X,
                                    float* __restrict__ Y,
                                    long long nwork) {
    long long t = (long long)blockIdx.x * blockDim.x + threadIdx.x;
    if (t >= nwork) return;
    const int e  = (int)(t >> 6);
    const int f4 = (int)(t & 63) << 2;
    const int r = rows[e];
    const int c = cols[e];
    const float v = vals[e];
    const float4 x = *(const float4*)(X + (long long)c * F_DIM + f4);
    float* y = Y + (long long)r * F_DIM + f4;
    atomicAdd(y + 0, v * x.x);
    atomicAdd(y + 1, v * x.y);
    atomicAdd(y + 2, v * x.z);
    atomicAdd(y + 3, v * x.w);
}

__global__ void leaky_relu_kernel(float* __restrict__ x, long long n) {
    long long i = (long long)blockIdx.x * blockDim.x + threadIdx.x;
    if (i >= n) return;
    float v = x[i];
    x[i] = v > 0.f ? v : 0.1f * v;
}

// out[0:n] holds raw U_s, out[n:2n] holds raw U_t; produce blended result in both.
__global__ void combine_kernel(float* __restrict__ out,
                               const float* __restrict__ bsu,
                               const float* __restrict__ btu,
                               long long n) {
    long long i = (long long)blockIdx.x * blockDim.x + threadIdx.x;
    if (i >= n) return;
    const int f = (int)(i & (F_DIM - 1));
    const float su = out[i] + bsu[f];
    const float tu = out[i + n] + btu[f];
    const float o = 0.5f * fmaxf(su, 0.f) + 0.5f * fmaxf(tu, 0.f);
    out[i] = o;
    out[i + n] = o;
}

// ---------------------------------------------------------------------------
extern "C" void kernel_launch(void* const* d_in, const int* in_sizes, int n_in,
                              void* d_out, int out_size, void* d_ws, size_t ws_size,
                              hipStream_t stream) {
    const float* src   = (const float*)d_in[0];
    const float* tgt   = (const float*)d_in[1];
    const int*   s_uv_r = (const int*)d_in[2];
    const int*   s_uv_c = (const int*)d_in[3];
    const float* s_uv_v = (const float*)d_in[4];
    const int*   s_vu_r = (const int*)d_in[5];
    const int*   s_vu_c = (const int*)d_in[6];
    const float* s_vu_v = (const float*)d_in[7];
    const int*   t_uv_r = (const int*)d_in[8];
    const int*   t_uv_c = (const int*)d_in[9];
    const float* t_uv_v = (const float*)d_in[10];
    const int*   t_vu_r = (const int*)d_in[11];
    const int*   t_vu_c = (const int*)d_in[12];
    const float* t_vu_v = (const float*)d_in[13];
    const float* W1  = (const float*)d_in[14];
    const float* W2  = (const float*)d_in[15];
    const float* W3  = (const float*)d_in[16];
    const float* W4  = (const float*)d_in[17];
    const float* Wsu = (const float*)d_in[18];
    const float* bsu = (const float*)d_in[19];
    const float* Wtu = (const float*)d_in[20];
    const float* btu = (const float*)d_in[21];

    float* out = (float*)d_out;
    const long long nU = (long long)N_USERS_C * F_DIM;  // 25,600,000
    const long long nI = (long long)N_ITEMS_C * F_DIM;  // 12,800,000

    // Workspace layout: R0 [users x F], R1 [items x F], R2 [items x F],
    // then 6 packed bf16 weight buffers.
    float* R0 = (float*)d_ws;
    float* R1 = R0 + nU;
    float* R2 = R1 + nI;
    __bf16* packBase = (__bf16*)(R2 + nI);
    const long long pack256 = 16LL * 8  * 1024;   // NT * Ksteps(8)  * 1024
    const long long pack512 = 16LL * 16 * 1024;   // NT * Ksteps(16) * 1024
    __bf16* P_W1  = packBase;
    __bf16* P_W2  = P_W1 + pack256;
    __bf16* P_W3  = P_W2 + pack256;
    __bf16* P_W4  = P_W3 + pack256;
    __bf16* P_Wsu = P_W4 + pack256;
    __bf16* P_Wtu = P_Wsu + pack512;

    const int NT = F_DIM / 16;                 // 16 N-tiles
    const int NtG = NT / NTW;                  // 4 wave-strips along N
    const int gU = (N_USERS_C / 16) * NtG;     // 25000 strips
    const int gI = (N_ITEMS_C / 16) * NtG;     // 12500 strips
    const long long spmmWork = (long long)NNZ_C * 64;
    const int spmmBlocks = (int)((spmmWork + 255) / 256);

    // Pre-pack all weights (tiny: <=256 waves each).
    pack_b_kernel<<<NT * 8,  32, 0, stream>>>(W1,  F_DIM, 8,  P_W1);
    pack_b_kernel<<<NT * 8,  32, 0, stream>>>(W2,  F_DIM, 8,  P_W2);
    pack_b_kernel<<<NT * 8,  32, 0, stream>>>(W3,  F_DIM, 8,  P_W3);
    pack_b_kernel<<<NT * 8,  32, 0, stream>>>(W4,  F_DIM, 8,  P_W4);
    pack_b_kernel<<<NT * 16, 32, 0, stream>>>(Wsu, F_DIM, 16, P_Wsu);
    pack_b_kernel<<<NT * 16, 32, 0, stream>>>(Wtu, F_DIM, 16, P_Wtu);

    for (int side = 0; side < 2; ++side) {
        const float* X     = side == 0 ? src  : tgt;
        const __bf16* Pa   = side == 0 ? P_W1 : P_W2;
        const __bf16* Pb   = side == 0 ? P_W3 : P_W4;
        const __bf16* Pcat = side == 0 ? P_Wsu : P_Wtu;
        const int* vu_r = side == 0 ? s_vu_r : t_vu_r;
        const int* vu_c = side == 0 ? s_vu_c : t_vu_c;
        const float* vu_v = side == 0 ? s_vu_v : t_vu_v;
        const int* uv_r = side == 0 ? s_uv_r : t_uv_r;
        const int* uv_c = side == 0 ? s_uv_c : t_uv_c;
        const float* uv_v = side == 0 ? s_uv_v : t_uv_v;
        float* U = out + side * nU;

        // 1) R0 = X @ Wa           [users x F]
        gemm_bf16x3_kernel<<<gU, 32, 0, stream>>>(X, F_DIM, F_DIM,
                                                  nullptr, 0, 0,
                                                  Pa, R0, F_DIM, NtG);
        // 2) R1 = leaky(spmm(vu, R0))   [items x F]
        hipMemsetAsync(R1, 0, (size_t)nI * sizeof(float), stream);
        spmm_scatter_kernel<<<spmmBlocks, 256, 0, stream>>>(vu_r, vu_c, vu_v, R0, R1, spmmWork);
        leaky_relu_kernel<<<(int)((nI + 255) / 256), 256, 0, stream>>>(R1, nI);
        // 3) R2 = R1 @ Wb          [items x F]
        gemm_bf16x3_kernel<<<gI, 32, 0, stream>>>(R1, F_DIM, F_DIM,
                                                  nullptr, 0, 0,
                                                  Pb, R2, F_DIM, NtG);
        // 4) R0 = leaky(spmm(uv, R2))   [users x F]   (old R0 already consumed)
        hipMemsetAsync(R0, 0, (size_t)nU * sizeof(float), stream);
        spmm_scatter_kernel<<<spmmBlocks, 256, 0, stream>>>(uv_r, uv_c, uv_v, R2, R0, spmmWork);
        leaky_relu_kernel<<<(int)((nU + 255) / 256), 256, 0, stream>>>(R0, nU);
        // 5) U = [R0, X] @ Wcat    (K = 512 via virtual concat), raw (bias in combine)
        gemm_bf16x3_kernel<<<gU, 32, 0, stream>>>(R0, F_DIM, F_DIM,
                                                  X, F_DIM, F_DIM,
                                                  Pcat, U, F_DIM, NtG);
    }

    // 6) out = 0.5*relu(U_s + bsu) + 0.5*relu(U_t + btu), written to both halves
    combine_kernel<<<(int)((nU + 255) / 256), 256, 0, stream>>>(out, bsu, btu, nU);
}
